// PyraformerSSModel_47210280517648
// MI455X (gfx1250) — matrix operand order repaired and checked by reading, not compiled, not run on hardware
//
#include <hip/hip_runtime.h>
#include <hip/hip_bf16.h>
#include <stdint.h>

typedef _Float16 h16;
typedef __attribute__((ext_vector_type(4)))  _Float16 h16x4;
typedef __attribute__((ext_vector_type(16))) _Float16 v16h;
typedef __attribute__((ext_vector_type(8)))  float    v8f;
typedef __attribute__((ext_vector_type(4)))  unsigned int u32x4;
typedef __attribute__((ext_vector_type(8)))  int      i32x8;
typedef __attribute__((ext_vector_type(4)))  int      i32x4;

#define WMMA(a, b, c) __builtin_amdgcn_wmma_f32_16x16x32_f16( \
    false, (a), false, (b), (short)0, (c), false, false)

#if defined(__has_builtin)
#if __has_builtin(__builtin_amdgcn_tensor_load_to_lds) && \
    __has_builtin(__builtin_amdgcn_s_wait_tensorcnt)
#define HAVE_TDM 1
#endif
#endif

// ---------------- WMMA fragment loaders (contiguous k = 16*hi + e, same for A and B) ----------------
__device__ __forceinline__ v16h fragA_nmajor(const h16* base, int rs) {
  int lane = threadIdx.x & 31;
  int l = lane & 15, hi = lane >> 4;
  v16h f;
#pragma unroll
  for (int e = 0; e < 16; ++e) {
    int kk = e + (hi << 4);
    f[e] = base[l * rs + kk];
  }
  return f;
}
__device__ __forceinline__ v16h fragB_nmajor(const h16* base, int rs) {
  int lane = threadIdx.x & 31;
  int l = lane & 15, hi = lane >> 4;
  v16h f;
#pragma unroll
  for (int e = 0; e < 16; ++e) f[e] = base[l * rs + e + (hi << 4)];
  return f;
}
__device__ __forceinline__ v16h fragB_kmajor(const h16* base, int rs) {
  int lane = threadIdx.x & 31;
  int l = lane & 15, hi = lane >> 4;
  v16h f;
#pragma unroll
  for (int e = 0; e < 16; ++e) f[e] = base[(e + (hi << 4)) * rs + l];
  return f;
}

__device__ __forceinline__ void st_h4(h16* dst, float4 f) {
  h16x4 hv;
  hv[0] = (h16)f.x; hv[1] = (h16)f.y; hv[2] = (h16)f.z; hv[3] = (h16)f.w;
  *(h16x4*)dst = hv;   // 8B aligned ds_store_b64
}

#ifdef HAVE_TDM
// 2D TDM load of an f32 tile into LDS at byte offset lds_off (one per wave).
__device__ __forceinline__ void tdm_load_2d_f32(unsigned lds_off, const float* g,
                                                unsigned tens_w, unsigned tens_h,
                                                unsigned tile_w, unsigned tile_h,
                                                unsigned stride0) {
  unsigned long long ga = (unsigned long long)(uintptr_t)g;
  u32x4 g0;
  g0[0] = 1u;                                       // count=1, user mode
  g0[1] = lds_off;                                  // lds_addr (bytes)
  g0[2] = (unsigned)(ga & 0xFFFFFFFFu);
  g0[3] = (unsigned)((ga >> 32) & 0x01FFFFFFu) | 0x80000000u;  // addr[56:32]|type=2
  i32x8 g1;
  g1[0] = (int)(2u << 16);                          // data_size = 4B
  g1[1] = (int)((tens_w & 0xFFFFu) << 16);
  g1[2] = (int)((tens_w >> 16) | ((tens_h & 0xFFFFu) << 16));
  g1[3] = (int)((tens_h >> 16) | (tile_w << 16));
  g1[4] = (int)tile_h;
  g1[5] = (int)stride0;
  g1[6] = 0;
  g1[7] = 0;
  i32x4 gz = {0, 0, 0, 0};
#if defined(__clang_major__) && (__clang_major__ >= 23)
  i32x8 gz8 = {0, 0, 0, 0, 0, 0, 0, 0};
  __builtin_amdgcn_tensor_load_to_lds(g0, g1, gz, gz, gz8, 0);
#else
  __builtin_amdgcn_tensor_load_to_lds(g0, g1, gz, gz, 0);
#endif
}
#endif

// ---------------- GEMM: C[z][Mb,N] = A[z][Mb,K] * W[N,K]^T (+bias,+act) ------
// Block tile 64(M) x 128(N), 4 waves, each 32x64 (8 WMMA / k-step).
// W tile arrives via TDM into f32 staging (overlapped with A loads).
// OOB tile rows are CLAMPED, not zeroed: an OOB A row only feeds an output row
// that is never stored (row m of D comes only from row m of A); same for B/cols.
#define GEMM_LDS_BYTES 31744
#define WSTAGE_OFF 15360
__global__ void gemm_f32_wmma(const float* __restrict__ A, size_t aBS,
                              const float* __restrict__ W,
                              const float* __restrict__ bias,
                              float* __restrict__ C, size_t cBS,
                              int Mb, int N, int K, int act) {
  extern __shared__ char smem[];
  h16* As = (h16*)smem;
  h16* Bs = (h16*)(smem + 5120);
  float* Wstage = (float*)(smem + WSTAGE_OFF);
  const int tid  = threadIdx.x;           // 128
  const int wave = tid >> 5;
  const int lane = tid & 31;
  const int tileM = blockIdx.y * 64;
  const int tileN = blockIdx.x * 128;
  A += (size_t)blockIdx.z * aBS;
  C += (size_t)blockIdx.z * cBS;
  const int wm = (wave >> 1) * 32;
  const int wn = (wave & 1) * 64;
  v8f acc[2][4];
#pragma unroll
  for (int i = 0; i < 2; ++i)
#pragma unroll
    for (int j = 0; j < 4; ++j)
      acc[i][j] = (v8f){0.f, 0.f, 0.f, 0.f, 0.f, 0.f, 0.f, 0.f};

  for (int k0 = 0; k0 < K; k0 += 32) {
#ifdef HAVE_TDM
    if (wave == 0)                         // issue DMA first, overlap with A loads
      tdm_load_2d_f32(WSTAGE_OFF, W + (size_t)tileN * (size_t)K + k0,
                      (unsigned)(K - k0), (unsigned)(N - tileN), 32u, 128u,
                      (unsigned)K);
#endif
    // A tile: 64x32, unconditional clamped b128 loads, then batched stores
    float4 fa[4];
#pragma unroll
    for (int j = 0; j < 4; ++j) {
      int c = tid + 128 * j;               // 512 float4 chunks
      int row = c >> 3, col4 = (c & 7) * 4;
      int m = tileM + row;
      if (m >= Mb) m = Mb - 1;             // clamp (dont-care rows)
      fa[j] = *(const float4*)(A + (size_t)m * (size_t)K + k0 + col4);
    }
#pragma unroll
    for (int j = 0; j < 4; ++j) {
      int c = tid + 128 * j;
      int row = c >> 3, col4 = (c & 7) * 4;
      st_h4(As + row * 40 + col4, fa[j]);
    }
    if (k0 + 32 < K) {
      int m = tileM + (tid >> 1);
      if (m >= Mb) m = Mb - 1;
      __builtin_prefetch(A + (size_t)m * (size_t)K + k0 + 32, 0, 0);
    }
#ifdef HAVE_TDM
    if (wave == 0) __builtin_amdgcn_s_wait_tensorcnt(0);
    __syncthreads();
    // convert staged f32 -> f16 Bs (TDM zero-fills OOB rows)
#pragma unroll
    for (int jj = 0; jj < 2; ++jj) {
      float4 fw[4];
#pragma unroll
      for (int j = 0; j < 4; ++j) {
        int c = tid + 128 * (jj * 4 + j);  // 1024 float4 chunks
        int row = c >> 3, col4 = (c & 7) * 4;
        fw[j] = *(const float4*)(Wstage + row * 32 + col4);
      }
#pragma unroll
      for (int j = 0; j < 4; ++j) {
        int c = tid + 128 * (jj * 4 + j);
        int row = c >> 3, col4 = (c & 7) * 4;
        st_h4(Bs + row * 40 + col4, fw[j]);
      }
    }
#else
#pragma unroll
    for (int jj = 0; jj < 2; ++jj) {
      float4 fw[4];
#pragma unroll
      for (int j = 0; j < 4; ++j) {
        int c = tid + 128 * (jj * 4 + j);
        int row = c >> 3, col4 = (c & 7) * 4;
        int n = tileN + row;
        if (n >= N) n = N - 1;             // clamp (dont-care cols)
        fw[j] = *(const float4*)(W + (size_t)n * (size_t)K + k0 + col4);
      }
#pragma unroll
      for (int j = 0; j < 4; ++j) {
        int c = tid + 128 * (jj * 4 + j);
        int row = c >> 3, col4 = (c & 7) * 4;
        st_h4(Bs + row * 40 + col4, fw[j]);
      }
    }
#endif
    __syncthreads();
    v16h a0 = fragA_nmajor(As + wm * 40, 40);
    v16h a1 = fragA_nmajor(As + (wm + 16) * 40, 40);
#pragma unroll
    for (int t = 0; t < 4; ++t) {
      v16h b = fragB_nmajor(Bs + (wn + 16 * t) * 40, 40);
      acc[0][t] = WMMA(a0, b, acc[0][t]);
      acc[1][t] = WMMA(a1, b, acc[1][t]);
    }
    __syncthreads();
  }
  const int l = lane & 15, hi = lane >> 4;
#pragma unroll
  for (int tm = 0; tm < 2; ++tm)
#pragma unroll
    for (int tn = 0; tn < 4; ++tn)
#pragma unroll
      for (int r = 0; r < 8; ++r) {
        int gm = tileM + wm + tm * 16 + hi * 8 + r;
        int gn = tileN + wn + tn * 16 + l;
        if (gm < Mb && gn < N) {
          float v = acc[tm][tn][r];
          if (bias) v += bias[gn];
          if (act == 1) v = 0.5f * v * (1.0f + erff(v * 0.70710678118f));
          C[(size_t)gm * (size_t)N + gn] = v;
        }
      }
}

// ---------------- Flash attention: 4 waves/block share one K/V tile ----------
// Each wave owns a 16-row Q tile; K/V (32 keys x 64) loaded cooperatively once
// per block -> 4x less global K/V traffic. Ps is wave-private (same-wave LDS
// ordering). Tail waves clamp to the last tile (recompute identical values).
__global__ void attn_flash(const float* __restrict__ q, const float* __restrict__ kbuf,
                           const float* __restrict__ vbuf,
                           const unsigned char* __restrict__ mask,
                           float* __restrict__ out) {
  const int S = 2720;
  const int h = blockIdx.y, b = blockIdx.z;
  const int tid = threadIdx.x;             // 128
  const int wave = tid >> 5;
  const int lane = tid & 31;
  const int l = lane & 15, hi = lane >> 4;
  int qt = blockIdx.x * 4 + wave;
  if (qt > 169) qt = 169;                  // clamp: duplicates are deterministic
  __shared__ __align__(16) h16 Qs[4 * 16 * 72];
  __shared__ __align__(16) h16 Ks[32 * 72];
  __shared__ __align__(16) h16 Vs[32 * 72];
  __shared__ __align__(16) h16 Ps[4 * 16 * 40];
  h16* Qw = Qs + wave * 16 * 72;
  h16* Pw = Ps + wave * 16 * 40;
  const size_t baseq = (size_t)(b * S) * 512 + (size_t)h * 64;
  {
    float4 fq[8];
#pragma unroll
    for (int j = 0; j < 8; ++j) {          // 16x64 floats = 256 chunks / 32 lanes
      int c = lane + 32 * j;
      int row = c >> 4, col4 = (c & 15) * 4;
      fq[j] = *(const float4*)(q + baseq + (size_t)(qt * 16 + row) * 512 + col4);
    }
#pragma unroll
    for (int j = 0; j < 8; ++j) {
      int c = lane + 32 * j;
      int row = c >> 4, col4 = (c & 15) * 4;
      fq[j].x *= 0.125f; fq[j].y *= 0.125f; fq[j].z *= 0.125f; fq[j].w *= 0.125f;
      st_h4(Qw + row * 72 + col4, fq[j]);
    }
  }
  v8f o0{}, o1{}, o2{}, o3{};
  float mrow[8], lrow[8];
#pragma unroll
  for (int r = 0; r < 8; ++r) { mrow[r] = -3.0e38f; lrow[r] = 0.f; }

  for (int kt = 0; kt < S; kt += 32) {
    {
      float4 kf[4], vf[4];
#pragma unroll
      for (int j = 0; j < 4; ++j) {        // 512 chunks / 128 threads
        int c = tid + 128 * j;
        int row = c >> 4, col4 = (c & 15) * 4;
        size_t goff = baseq + (size_t)(kt + row) * 512 + col4;
        kf[j] = *(const float4*)(kbuf + goff);
        vf[j] = *(const float4*)(vbuf + goff);
      }
#pragma unroll
      for (int j = 0; j < 4; ++j) {
        int c = tid + 128 * j;
        int row = c >> 4, col4 = (c & 15) * 4;
        st_h4(Ks + row * 72 + col4, kf[j]);
        st_h4(Vs + row * 72 + col4, vf[j]);
      }
    }
    __syncthreads();
    v16h aq0 = fragA_nmajor(Qw, 72);
    v16h aq1 = fragA_nmajor(Qw + 32, 72);
    v8f s0 = (v8f){0.f, 0.f, 0.f, 0.f, 0.f, 0.f, 0.f, 0.f};
    v8f s1 = s0;
    s0 = WMMA(aq0, fragB_nmajor(Ks, 72), s0);
    s0 = WMMA(aq1, fragB_nmajor(Ks + 32, 72), s0);
    s1 = WMMA(aq0, fragB_nmajor(Ks + 16 * 72, 72), s1);
    s1 = WMMA(aq1, fragB_nmajor(Ks + 16 * 72 + 32, 72), s1);
    const int qrow0 = qt * 16 + hi * 8;
#pragma unroll
    for (int r = 0; r < 8; ++r) {
      size_t mro = (size_t)(qrow0 + r) * (size_t)S;
      float a = s0[r], bb = s1[r];
      if (mask[mro + kt + l])      a  = -1e9f;
      if (mask[mro + kt + 16 + l]) bb = -1e9f;
      float tmax = fmaxf(a, bb);
#pragma unroll
      for (int off = 1; off < 16; off <<= 1)
        tmax = fmaxf(tmax, __shfl_xor(tmax, off, 32));
      float mnew = fmaxf(mrow[r], tmax);
      float corr = expf(mrow[r] - mnew);
      float e0 = expf(a - mnew), e1 = expf(bb - mnew);
      float rs = e0 + e1;
#pragma unroll
      for (int off = 1; off < 16; off <<= 1) rs += __shfl_xor(rs, off, 32);
      lrow[r] = lrow[r] * corr + rs;
      mrow[r] = mnew;
      o0[r] *= corr; o1[r] *= corr; o2[r] *= corr; o3[r] *= corr;
      Pw[(hi * 8 + r) * 40 + l]      = (h16)e0;
      Pw[(hi * 8 + r) * 40 + 16 + l] = (h16)e1;
    }
    // Ps store->load is same-wave LDS: in order, no block barrier needed
    v16h ap = fragA_nmajor(Pw, 40);
    o0 = WMMA(ap, fragB_kmajor(Vs + 0, 72), o0);
    o1 = WMMA(ap, fragB_kmajor(Vs + 16, 72), o1);
    o2 = WMMA(ap, fragB_kmajor(Vs + 32, 72), o2);
    o3 = WMMA(ap, fragB_kmajor(Vs + 48, 72), o3);
    __syncthreads();
  }
#pragma unroll
  for (int r = 0; r < 8; ++r) {
    int qrow = qt * 16 + hi * 8 + r;
    float inv = 1.0f / lrow[r];
    size_t off = baseq + (size_t)qrow * 512;
    out[off + 0  + l] = o0[r] * inv;
    out[off + 16 + l] = o1[r] * inv;
    out[off + 32 + l] = o2[r] * inv;
    out[off + 48 + l] = o3[r] * inv;
  }
}

// ---------------- Embedding: cov linear + circular conv3 + sinusoidal PE -----
__global__ void embed_kernel(const float* __restrict__ data,
                             const float* __restrict__ cw, const float* __restrict__ cb,
                             const float* __restrict__ w3, float* __restrict__ xfull) {
  const int L = 2048;
  int bid = blockIdx.x;
  int b = bid >> 11, l = bid & 2047;
  const float* row = data + (size_t)(b * L + l) * 5;
  float c0 = row[1], c1 = row[2], c2 = row[3];
  float c3 = row[4] * 0.01f - 0.5f;
  int lm = (l + L - 1) & (L - 1);
  int lp = (l + 1) & (L - 1);
  float tm = data[(size_t)(b * L + lm) * 5];
  float t0 = row[0];
  float tp = data[(size_t)(b * L + lp) * 5];
  size_t ob = (size_t)(b * 2720 + l) * 512;
  for (int d = threadIdx.x; d < 512; d += 128) {
    float ce = cw[d * 4] * c0 + cw[d * 4 + 1] * c1 + cw[d * 4 + 2] * c2 +
               cw[d * 4 + 3] * c3 + cb[d];
    float de = w3[d * 3] * tm + w3[d * 3 + 1] * t0 + w3[d * 3 + 2] * tp;
    float pv = powf(10000.0f, (float)(2 * (d >> 1)) * (1.0f / 512.0f));
    float ang = (float)l / pv;
    float pe = ((d & 1) == 0) ? sinf(ang) : cosf(ang);
    xfull[ob + d] = ce + de + pe;
  }
}

// ---- pack conv weight [CO][CI][4] -> [CO][j*64+ci] so conv == GEMM(K=256) ---
__global__ void pack_conv_w(const float* __restrict__ src, float* __restrict__ wp) {
  int i = blockIdx.x * 256 + threadIdx.x;
  if (i < 64 * 64 * 4) {
    int co = i >> 8, rem = i & 255, ci = rem >> 2, j = rem & 3;
    wp[co * 256 + j * 64 + ci] = src[i];
  }
}

// ---------------- BatchNorm stats (train mode: over B*len per channel) -------
__global__ void bn_stats(const float* __restrict__ x, float* __restrict__ stats, int M) {
  int c = blockIdx.x, tid = threadIdx.x;
  float s = 0.f, s2 = 0.f;
  for (int m = tid; m < M; m += 256) {
    float v = x[(size_t)m * 64 + c];
    s += v; s2 += v * v;
  }
  __shared__ float r1[256], r2[256];
  r1[tid] = s; r2[tid] = s2;
  __syncthreads();
  for (int o = 128; o > 0; o >>= 1) {
    if (tid < o) { r1[tid] += r1[tid + o]; r2[tid] += r2[tid + o]; }
    __syncthreads();
  }
  if (tid == 0) {
    float mu = r1[0] / (float)M;
    stats[c * 2] = mu;
    stats[c * 2 + 1] = r2[0] / (float)M - mu * mu;
  }
}

__global__ void bn_elu_scatter(const float* __restrict__ raw, const float* __restrict__ stats,
                               const float* __restrict__ g, const float* __restrict__ bb,
                               float* __restrict__ pcat, int lenShift, int off, int M) {
  int i = blockIdx.x * 256 + threadIdx.x;
  if (i >= M * 64) return;
  int m = i >> 6, c = i & 63;
  float mu = stats[c * 2], var = stats[c * 2 + 1];
  float v = (raw[i] - mu) * rsqrtf(var + 1e-5f) * g[c] + bb[c];
  v = v > 0.f ? v : expm1f(v);
  int bi = m >> lenShift, t = m & ((1 << lenShift) - 1);
  pcat[(size_t)(bi * 672 + off + t) * 64 + c] = v;
}

// ---------------- LayerNorm (optional residual add), D=512 -------------------
__global__ void ln_kernel(const float* __restrict__ y, float* __restrict__ x,
                          const float* __restrict__ g, const float* __restrict__ bta,
                          float eps) {
  int row = blockIdx.x, tid = threadIdx.x;   // 256 threads
  size_t base = (size_t)row * 512;
  float v0 = x[base + tid]       + (y ? y[base + tid]       : 0.f);
  float v1 = x[base + 256 + tid] + (y ? y[base + 256 + tid] : 0.f);
  __shared__ float r1[256], r2[256];
  r1[tid] = v0 + v1;
  r2[tid] = v0 * v0 + v1 * v1;
  __syncthreads();
  for (int o = 128; o > 0; o >>= 1) {
    if (tid < o) { r1[tid] += r1[tid + o]; r2[tid] += r2[tid + o]; }
    __syncthreads();
  }
  __shared__ float smu, srstd;
  if (tid == 0) {
    float m = r1[0] * (1.0f / 512.0f);
    float var = r2[0] * (1.0f / 512.0f) - m * m;
    smu = m; srstd = rsqrtf(var + eps);
  }
  __syncthreads();
  x[base + tid]       = (v0 - smu) * srstd * g[tid]       + bta[tid];
  x[base + 256 + tid] = (v1 - smu) * srstd * g[256 + tid] + bta[256 + tid];
}

// ---------------- gather 4 scales + two softplus heads -----------------------
__global__ void heads_kernel(const float* __restrict__ xfull, const int* __restrict__ gidx,
                             const float* __restrict__ mw, const float* __restrict__ vw,
                             float* __restrict__ out) {
  const int L = 2048;
  int bid = blockIdx.x;
  int b = bid >> 11, l = bid & 2047;
  int tid = threadIdx.x;
  int rows[4];
#pragma unroll
  for (int s = 0; s < 4; ++s) rows[s] = gidx[l * 4 + s];
  float pm = 0.f, pv = 0.f;
  for (int e = tid; e < 2048; e += 256) {
    int s = e >> 9, dd = e & 511;
    float v = xfull[(size_t)(b * 2720 + rows[s]) * 512 + dd];
    pm += v * mw[e];
    pv += v * vw[e];
  }
  __shared__ float r1[256], r2[256];
  r1[tid] = pm; r2[tid] = pv;
  __syncthreads();
  for (int o = 128; o > 0; o >>= 1) {
    if (tid < o) { r1[tid] += r1[tid + o]; r2[tid] += r2[tid + o]; }
    __syncthreads();
  }
  if (tid == 0) {
    float sm = r1[0], sv = r2[0];
    out[(size_t)b * L + l]                 = sm > 20.f ? sm : log1pf(expf(sm));
    out[(size_t)2 * L + (size_t)b * L + l] = sv > 20.f ? sv : log1pf(expf(sv));
  }
}

// =============================================================================
extern "C" void kernel_launch(void* const* d_in, const int* in_sizes, int n_in,
                              void* d_out, int out_size, void* d_ws, size_t ws_size,
                              hipStream_t stream) {
  (void)in_sizes; (void)n_in; (void)out_size; (void)ws_size;
  const float* data   = (const float*)d_in[0];
  const float* cov_w  = (const float*)d_in[1];
  const float* cov_b  = (const float*)d_in[2];
  const float* demb_w = (const float*)d_in[3];
  const float* down_w = (const float*)d_in[4];
  const float* down_b = (const float*)d_in[5];
  const float* conv_w[3] = {(const float*)d_in[6],  (const float*)d_in[7],  (const float*)d_in[8]};
  const float* conv_b[3] = {(const float*)d_in[9],  (const float*)d_in[10], (const float*)d_in[11]};
  const float* bn_g[3]   = {(const float*)d_in[12], (const float*)d_in[13], (const float*)d_in[14]};
  const float* bn_b[3]   = {(const float*)d_in[15], (const float*)d_in[16], (const float*)d_in[17]};
  const float* up_w   = (const float*)d_in[18];
  const float* up_b   = (const float*)d_in[19];
  const float* norm_g = (const float*)d_in[20];
  const float* norm_b = (const float*)d_in[21];
  const float* mean_w = (const float*)d_in[74];
  const float* var_w  = (const float*)d_in[75];
  const unsigned char* mask = (const unsigned char*)d_in[76];
  const int* gidx = (const int*)d_in[77];

  float* ws = (float*)d_ws;
  const size_t SZ = (size_t)5440 * 512;
  float* xfull = ws;
  float* qb  = xfull + SZ;
  float* kb  = qb + SZ;
  float* vb  = kb + SZ;
  float* ab  = vb + SZ;
  float* yb  = ab + SZ;
  float* hb  = yb + SZ;                        // ffn hidden chunk (1360*2048)
  float* tmpd  = hb + SZ;                      // [B][2048,64]
  float* craw  = tmpd + (size_t)4096 * 64;     // [B][len,64] max 1024 rows
  float* pcat  = craw + (size_t)1024 * 64;     // [B][672,64]
  float* wpack = pcat + (size_t)1344 * 64;     // [64,256]
  float* stats = wpack + 16384;                // [64,2]

  auto gemm = [&](const float* A, size_t aBS, const float* W, const float* bias,
                  float* C, size_t cBS, int Mb, int N, int K, int act, int nb) {
    dim3 grid((N + 127) / 128, (Mb + 63) / 64, nb);
    gemm_f32_wmma<<<grid, 128, GEMM_LDS_BYTES, stream>>>(A, aBS, W, bias, C, cBS,
                                                         Mb, N, K, act);
  };
  const size_t XS = (size_t)2720 * 512;

  // 1) embedding -> xfull rows [b][0..2047]
  embed_kernel<<<4096, 128, 0, stream>>>(data, cov_w, cov_b, demb_w, xfull);
  // 2) down projection per batch
  gemm(xfull, XS, down_w, down_b, tmpd, (size_t)2048 * 64, 2048, 64, 512, 0, 2);
  // 3) pyramid convs (GEMM K=256) + BN(train) + ELU
  const int lens[3] = {512, 128, 32};
  const int shifts[3] = {9, 7, 5};
  const int offs[3] = {0, 512, 640};
  for (int lvl = 0; lvl < 3; ++lvl) {
    pack_conv_w<<<64, 256, 0, stream>>>(conv_w[lvl], wpack);
    const float* Ain; size_t aBS;
    if (lvl == 0) { Ain = tmpd; aBS = (size_t)2048 * 64; }
    else { Ain = pcat + (size_t)offs[lvl - 1] * 64; aBS = (size_t)672 * 64; }
    int len = lens[lvl];
    gemm(Ain, aBS, wpack, conv_b[lvl], craw, (size_t)len * 64, len, 64, 256, 0, 2);
    bn_stats<<<64, 256, 0, stream>>>(craw, stats, 2 * len);
    bn_elu_scatter<<<(2 * len * 64 + 255) / 256, 256, 0, stream>>>(
        craw, stats, bn_g[lvl], bn_b[lvl], pcat, shifts[lvl], offs[lvl], 2 * len);
  }
  // 4) up projection -> xfull rows [b][2048..2719]
  gemm(pcat, (size_t)672 * 64, up_w, up_b, xfull + (size_t)2048 * 512, XS,
       672, 512, 64, 0, 2);
  // 5) initial LayerNorm
  ln_kernel<<<5440, 256, 0, stream>>>(nullptr, xfull, norm_g, norm_b, 1e-5f);
  // 6) encoder layers
  for (int li = 0; li < 4; ++li) {
    const float* wq   = (const float*)d_in[22 + 13 * li + 0];
    const float* wk   = (const float*)d_in[22 + 13 * li + 1];
    const float* wv   = (const float*)d_in[22 + 13 * li + 2];
    const float* fc_w = (const float*)d_in[22 + 13 * li + 3];
    const float* fc_b = (const float*)d_in[22 + 13 * li + 4];
    const float* l1g  = (const float*)d_in[22 + 13 * li + 5];
    const float* l1b  = (const float*)d_in[22 + 13 * li + 6];
    const float* w1   = (const float*)d_in[22 + 13 * li + 7];
    const float* b1   = (const float*)d_in[22 + 13 * li + 8];
    const float* w2   = (const float*)d_in[22 + 13 * li + 9];
    const float* b2   = (const float*)d_in[22 + 13 * li + 10];
    const float* l2g  = (const float*)d_in[22 + 13 * li + 11];
    const float* l2b  = (const float*)d_in[22 + 13 * li + 12];
    gemm(xfull, 0, wq, nullptr, qb, 0, 5440, 512, 512, 0, 1);
    gemm(xfull, 0, wk, nullptr, kb, 0, 5440, 512, 512, 0, 1);
    gemm(xfull, 0, wv, nullptr, vb, 0, 5440, 512, 512, 0, 1);
    dim3 ag(43, 8, 2);                       // 4 q-tiles per block (4 waves)
    attn_flash<<<ag, 128, 0, stream>>>(qb, kb, vb, mask, ab);
    gemm(ab, 0, fc_w, fc_b, yb, 0, 5440, 512, 512, 0, 1);
    ln_kernel<<<5440, 256, 0, stream>>>(yb, xfull, l1g, l1b, 1e-6f);
    for (int c = 0; c < 4; ++c) {
      const float* xa = xfull + (size_t)c * 1360 * 512;
      gemm(xa, 0, w1, b1, hb, 0, 1360, 2048, 512, 1, 1);
      gemm(hb, 0, w2, b2, yb + (size_t)c * 1360 * 512, 0, 1360, 512, 2048, 0, 1);
    }
    ln_kernel<<<5440, 256, 0, stream>>>(yb, xfull, l2g, l2b, 1e-6f);
  }
  // 7) gather + softplus heads -> d_out (mean then var)
  heads_kernel<<<4096, 256, 0, stream>>>(xfull, gidx, mean_w, var_w, (float*)d_out);
}
